// RelAttSkew_11201274708207
// MI455X (gfx1250) — compile-verified, matrix operand order checked by reading, not run on hardware
//
#include <hip/hip_runtime.h>
#include <hip/hip_bf16.h>

// RelAttSkew on MI455X (gfx1250): all GEMMs via v_wmma_f32_16x16x32_f16.
// B=4, S=1024, D=512, H=8, dh=64. Output-store bound (~411MB of f32 out),
// so f16 WMMA with f32 accumulate is far past the compute roofline.
// V is stored transposed [B,H,dh,S] so every WMMA B-fragment load in the
// hot path is a contiguous 16B vector load (global_load_b128).

typedef __attribute__((ext_vector_type(16))) _Float16 v16h;
typedef __attribute__((ext_vector_type(8)))  _Float16 v8h;
typedef __attribute__((ext_vector_type(8)))  float    v8f;
typedef __attribute__((ext_vector_type(4)))  float    f32x4;

#define WMMA16(A, B, C) __builtin_amdgcn_wmma_f32_16x16x32_f16( \
    false, (A), false, (B), (short)0, (C), false, false)

// ---- Fragment loaders (CDNA5 16-bit A/B layouts, ISA 7.12.2) -------------
// A 16x32 f16: lane l -> row = row0 + (l&15); kb = k0 + (l>>4)*8;
// halves[0..7] = src[row, kb..kb+7], halves[8..15] = src[row, kb+16..kb+23].
// A B^T-fragment from a row-major [N,K] matrix uses the identical pattern
// with "row" = n. Caller passes already-offset row and kb.

__device__ __forceinline__ v16h frag_row_f32(const float* __restrict__ src,
                                             int ld, int row, int kb) {
  const float* p = src + (size_t)row * ld + kb;
  f32x4 a0 = *(const f32x4*)(p);
  f32x4 a1 = *(const f32x4*)(p + 4);
  f32x4 a2 = *(const f32x4*)(p + 16);
  f32x4 a3 = *(const f32x4*)(p + 20);
  v16h f;
#pragma unroll
  for (int i = 0; i < 4; i++) {
    f[i]      = (_Float16)a0[i];
    f[4 + i]  = (_Float16)a1[i];
    f[8 + i]  = (_Float16)a2[i];
    f[12 + i] = (_Float16)a3[i];
  }
  return f;
}

__device__ __forceinline__ v16h frag_row_f16(const _Float16* __restrict__ src,
                                             int ld, int row, int kb) {
  const _Float16* p = src + (size_t)row * ld + kb;
  v8h a0 = *(const v8h*)(p);
  v8h a1 = *(const v8h*)(p + 16);
  v16h f;
#pragma unroll
  for (int i = 0; i < 8; i++) { f[i] = a0[i]; f[8 + i] = a1[i]; }
  return f;
}

// ---- Kernel 1: Q/K projection  out_f16[B,H,S,dh] = (X @ W^T + b) ----------
// X [4096,512] f32, W [512,512] f32. One 16x16 tile per wave.
__global__ void proj_qkv_kernel(const float* __restrict__ X,
                                const float* __restrict__ W,
                                const float* __restrict__ bias,
                                _Float16* __restrict__ out) {
  int l  = threadIdx.x & 31;
  int lr = l & 15, lk = (l >> 4) << 3;
  int tile = blockIdx.x * 8 + (threadIdx.x >> 5);  // 8192 tiles: 256m x 32n
  int m0 = (tile >> 5) << 4;
  int n0 = (tile & 31) << 4;

  v8f acc = {};
#pragma unroll 4
  for (int k0 = 0; k0 < 512; k0 += 32) {
    v16h a  = frag_row_f32(X, 512, m0 + lr, k0 + lk);
    v16h bf = frag_row_f32(W, 512, n0 + lr, k0 + lk);  // B^T of W
    acc = WMMA16(a, bf, acc);
  }
  int n = n0 + lr;
  float bv = bias[n];
  int h = n >> 6, dd = n & 63;
#pragma unroll
  for (int r = 0; r < 8; r++) {
    int m = m0 + lk + r;
    int b = m >> 10, s = m & 1023;
    out[(((size_t)(b * 8 + h) << 10) + s) * 64 + dd] = (_Float16)(acc[r] + bv);
  }
}

// ---- Kernel 1b: V projection, stored TRANSPOSED [B,H,dh,S] ----------------
// Each lane packs its 8 accumulator rows (consecutive s) into one v8h and
// writes a single 16B global_store_b128.
__global__ void proj_vT_kernel(const float* __restrict__ X,
                               const float* __restrict__ W,
                               const float* __restrict__ bias,
                               _Float16* __restrict__ outT) {
  int l  = threadIdx.x & 31;
  int lr = l & 15, lk = (l >> 4) << 3;
  int tile = blockIdx.x * 8 + (threadIdx.x >> 5);
  int m0 = (tile >> 5) << 4;
  int n0 = (tile & 31) << 4;

  v8f acc = {};
#pragma unroll 4
  for (int k0 = 0; k0 < 512; k0 += 32) {
    v16h a  = frag_row_f32(X, 512, m0 + lr, k0 + lk);
    v16h bf = frag_row_f32(W, 512, n0 + lr, k0 + lk);
    acc = WMMA16(a, bf, acc);
  }
  int n = n0 + lr;
  float bv = bias[n];
  int h = n >> 6, dd = n & 63;
  int m = m0 + lk;                 // 8 consecutive rows, same batch
  int b = m >> 10, s = m & 1023;   // s multiple of 8 -> 16B aligned store
  v8h pk;
#pragma unroll
  for (int r = 0; r < 8; r++) pk[r] = (_Float16)(acc[r] + bv);
  *(v8h*)(outT + (((size_t)(b * 8 + h) * 64 + dd) << 10) + s) = pk;
}

// ---- Kernel 2: rel = pos_emb @ Wp^T + bp, stored f16 padded to 2064 rows --
__global__ void rel_kernel(const float* __restrict__ pos_emb,
                           const float* __restrict__ Wp,
                           const float* __restrict__ bp,
                           _Float16* __restrict__ rel16) {
  int l  = threadIdx.x & 31;
  int lr = l & 15, lk = (l >> 4) << 3;
  int tile = blockIdx.x * 8 + (threadIdx.x >> 5);  // 129 m-tiles x 4 n-tiles
  if (tile >= 516) return;
  int m0 = (tile >> 2) << 4;
  int n0 = (tile & 3) << 4;

  v8f acc = {};
#pragma unroll
  for (int k0 = 0; k0 < 64; k0 += 32) {
    int row = m0 + lr; if (row > 2046) row = 2046;   // clamp OOB loads
    v16h a  = frag_row_f32(pos_emb, 64, row, k0 + lk);
    v16h bf = frag_row_f32(Wp, 64, n0 + lr, k0 + lk);
    acc = WMMA16(a, bf, acc);
  }
  int n = n0 + lr;
  float bv = bp[n];
#pragma unroll
  for (int r = 0; r < 8; r++) {
    int m = m0 + lk + r;
    float v = (m < 2047) ? (acc[r] + bv) : 0.0f;    // zero pad rows 2047..2063
    rel16[(size_t)m * 64 + n] = (_Float16)v;
  }
}

// ---- Kernel 3: fused attention row-block ----------------------------------
// One workgroup = (b, h, 16 rows of s). 256 threads = 8 waves.
// LDS: score[16][1024] f32 | raw[16][1056] f32 (reused as attn16) | red[256]
__global__ void attn_kernel(const _Float16* __restrict__ q16,
                            const _Float16* __restrict__ k16,
                            const _Float16* __restrict__ v16T,
                            const _Float16* __restrict__ rel16,
                            float* __restrict__ attn_o,
                            float* __restrict__ cont_o,
                            float* __restrict__ pos_o,
                            _Float16* __restrict__ ctx16) {
  extern __shared__ float smem[];
  float* score = smem;                          // 16*1024 f32
  float* raw   = smem + 16 * 1024;              // 16*1056 f32
  float* red   = smem + 16 * 1024 + 16 * 1056;  // 256 f32
  _Float16* attn16 = (_Float16*)raw;            // overlay after raw consumed

  int bid = blockIdx.x;
  int s0 = (bid & 63) << 4;
  int h  = (bid >> 6) & 7;
  int b  = bid >> 9;
  const size_t head = ((size_t)b * 8 + h) << 10;       // (b*H+h)*S
  const _Float16* qh  = q16  + head * 64;
  const _Float16* kh  = k16  + head * 64;
  const _Float16* vhT = v16T + head * 64;              // [64][1024]

  int tid = threadIdx.x;
  int wave = tid >> 5;
  int l = tid & 31;
  int lr = l & 15, lk = (l >> 4) << 3;

  // q fragments are invariant over all score tiles: load once.
  v16h aq0 = frag_row_f16(qh, 64, s0 + lr, 0  + lk);
  v16h aq1 = frag_row_f16(qh, 64, s0 + lr, 32 + lk);

  // --- content_score tiles: q_blk[16x64] @ K^T -----------------------------
  for (int tt = wave; tt < 64; tt += 8) {
    int t0 = tt << 4;
    v8f acc = {};
    acc = WMMA16(aq0, frag_row_f16(kh, 64, t0 + lr, 0  + lk), acc);
    acc = WMMA16(aq1, frag_row_f16(kh, 64, t0 + lr, 32 + lk), acc);
    int t = t0 + lr;
#pragma unroll
    for (int r = 0; r < 8; r++) {
      int srel = lk + r;
      float v = acc[r];
      cont_o[((head + s0 + srel) << 10) + t] = v;
      score[(srel << 10) + t] = v;
    }
  }

  // --- raw pos band: q_blk @ rel^T over r in [1008-s0, 1008-s0+1056) -------
  int rbase = 1008 - s0;
  for (int ct = wave; ct < 66; ct += 8) {
    int r0 = rbase + (ct << 4);
    v8f acc = {};
    acc = WMMA16(aq0, frag_row_f16(rel16, 64, r0 + lr, 0  + lk), acc);
    acc = WMMA16(aq1, frag_row_f16(rel16, 64, r0 + lr, 32 + lk), acc);
    int c = (ct << 4) + lr;
#pragma unroll
    for (int r = 0; r < 8; r++) raw[(lk + r) * 1056 + c] = acc[r];
  }
  __syncthreads();

  // --- skew shift: pos[s,t] = raw[s, t - s + S-1 - rbase] = raw[s, t+15-srel]
  for (int idx = tid; idx < 16 * 1024; idx += 256) {
    int srel = idx >> 10, t = idx & 1023;
    float pv = raw[srel * 1056 + (t + 15 - srel)];
    pos_o[((head + s0 + srel) << 10) + t] = pv;
    score[idx] += pv;
  }
  __syncthreads();

  // --- softmax over t (scale by 1/sqrt(D)), 16 threads per row -------------
  {
    const float sc = 0.04419417382415922f;  // 1/sqrt(512)
    int row = tid >> 4, sub = tid & 15;
    float* sr = score + (row << 10);
    float mx = -1e30f;
    for (int j = sub; j < 1024; j += 16) mx = fmaxf(mx, sr[j] * sc);
    red[(row << 4) + sub] = mx;
    __syncthreads();
    float rmx = -1e30f;
#pragma unroll
    for (int j = 0; j < 16; j++) rmx = fmaxf(rmx, red[(row << 4) + j]);
    __syncthreads();
    float lsum = 0.0f;
    for (int j = sub; j < 1024; j += 16) {
      float e = __expf(sr[j] * sc - rmx);
      sr[j] = e;
      lsum += e;
    }
    red[(row << 4) + sub] = lsum;
    __syncthreads();
    float rsum = 0.0f;
#pragma unroll
    for (int j = 0; j < 16; j++) rsum += red[(row << 4) + j];
    float inv = 1.0f / rsum;
    for (int j = sub; j < 1024; j += 16) {
      float a = sr[j] * inv;
      attn_o[((head + s0 + row) << 10) + j] = a;
      attn16[(row << 10) + j] = (_Float16)a;   // overlays raw (done with it)
    }
  }
  __syncthreads();   // attn16 visible; score region now free for partials

  // --- ctx = attn[16x1024] @ V[1024x64] ------------------------------------
  // 8 waves: wave&3 owns 16 output cols, wave>>2 owns half the K range.
  // V^T layout makes the B fragment a contiguous row load.
  {
    int d0 = (wave & 3) << 4;
    int kstart = (wave >> 2) << 9;              // 0 or 512
    v8f acc = {};
    for (int k0 = kstart; k0 < kstart + 512; k0 += 32) {
      v16h a  = frag_row_f16(attn16, 1024, lr, k0 + lk);       // LDS
      v16h bf = frag_row_f16(vhT, 1024, d0 + lr, k0 + lk);     // contiguous
      acc = WMMA16(a, bf, acc);
    }
    if (wave >= 4) {                            // stash partial in free LDS
      float* part = score + ((wave & 3) << 8);
#pragma unroll
      for (int r = 0; r < 8; r++) part[(l << 3) + r] = acc[r];
    }
    __syncthreads();
    if (wave < 4) {
      const float* part = score + (wave << 8);
#pragma unroll
      for (int r = 0; r < 8; r++) acc[r] += part[(l << 3) + r];
      int d = d0 + lr;
#pragma unroll
      for (int r = 0; r < 8; r++) {
        int srel = lk + r;
        ctx16[((size_t)b * 1024 + s0 + srel) * 512 + (h << 6) + d] =
            (_Float16)acc[r];
      }
    }
  }
}

// ---- Kernel 4: out = ctx @ Wo^T + bo --------------------------------------
__global__ void outproj_kernel(const _Float16* __restrict__ ctx16,
                               const float* __restrict__ Wo,
                               const float* __restrict__ bo,
                               float* __restrict__ out) {
  int l  = threadIdx.x & 31;
  int lr = l & 15, lk = (l >> 4) << 3;
  int tile = blockIdx.x * 8 + (threadIdx.x >> 5);
  int m0 = (tile >> 5) << 4;
  int n0 = (tile & 31) << 4;

  v8f acc = {};
#pragma unroll 4
  for (int k0 = 0; k0 < 512; k0 += 32) {
    v16h a  = frag_row_f16(ctx16, 512, m0 + lr, k0 + lk);
    v16h bf = frag_row_f32(Wo, 512, n0 + lr, k0 + lk);
    acc = WMMA16(a, bf, acc);
  }
  int n = n0 + lr;
  float bv = bo[n];
#pragma unroll
  for (int r = 0; r < 8; r++) {
    int m = m0 + lk + r;
    out[(size_t)m * 512 + n] = acc[r] + bv;
  }
}

extern "C" void kernel_launch(void* const* d_in, const int* in_sizes, int n_in,
                              void* d_out, int out_size, void* d_ws, size_t ws_size,
                              hipStream_t stream) {
  const float* query   = (const float*)d_in[0];
  const float* key     = (const float*)d_in[1];
  const float* value   = (const float*)d_in[2];
  const float* pos_emb = (const float*)d_in[3];
  const float* Wq = (const float*)d_in[4];
  const float* bq = (const float*)d_in[5];
  const float* Wk = (const float*)d_in[6];
  const float* bk = (const float*)d_in[7];
  const float* Wv = (const float*)d_in[8];
  const float* bv = (const float*)d_in[9];
  const float* Wo = (const float*)d_in[10];
  const float* bo = (const float*)d_in[11];
  const float* Wp = (const float*)d_in[12];
  const float* bp = (const float*)d_in[13];

  const size_t QKV = (size_t)4 * 1024 * 512;           // 2,097,152 elements
  _Float16* q16   = (_Float16*)d_ws;
  _Float16* k16   = q16 + QKV;
  _Float16* v16T  = k16 + QKV;                         // [B,H,dh,S]
  _Float16* ctx16 = v16T + QKV;
  _Float16* rel16 = ctx16 + QKV;                       // 2064*64 halfs

  float* out_o  = (float*)d_out;                       // [4,1024,512]
  float* attn_o = out_o + QKV;                         // [4,8,1024,1024]
  float* cont_o = attn_o + (size_t)33554432;
  float* pos_o  = cont_o + (size_t)33554432;

  proj_qkv_kernel<<<1024, 256, 0, stream>>>(query, Wq, bq, q16);
  proj_qkv_kernel<<<1024, 256, 0, stream>>>(key,   Wk, bk, k16);
  proj_vT_kernel<<<1024, 256, 0, stream>>>(value, Wv, bv, v16T);
  rel_kernel<<<65, 256, 0, stream>>>(pos_emb, Wp, bp, rel16);

  const size_t smem = (size_t)(16 * 1024 + 16 * 1056 + 256) * sizeof(float); // 134,144 B
  (void)hipFuncSetAttribute(reinterpret_cast<const void*>(attn_kernel),
                            hipFuncAttributeMaxDynamicSharedMemorySize, (int)smem);
  attn_kernel<<<2048, 256, smem, stream>>>(q16, k16, v16T, rel16,
                                           attn_o, cont_o, pos_o, ctx16);

  outproj_kernel<<<1024, 256, 0, stream>>>(ctx16, Wo, bo, out_o);
}
